// VAE_16552803959308
// MI455X (gfx1250) — matrix operand order repaired and checked by reading, not compiled, not run on hardware
//
#include <hip/hip_runtime.h>
#include <hip/hip_bf16.h>
#include <math.h>

typedef __attribute__((ext_vector_type(16))) _Float16 v16h;
typedef __attribute__((ext_vector_type(8)))  float    v8f;

#define BT 2048
#define TT 512
#define DD 8
#define HH 64
#define LL 16
#define PP 2

// output layout: recon | mu | log_var | num | mass (flat, f32)
#define MU_OFF   ((size_t)BT * TT * DD)
#define LV_OFF   (MU_OFF + (size_t)BT * LL)
#define NUM_OFF  (LV_OFF + (size_t)BT * LL)
#define MASS_OFF (NUM_OFF + (size_t)BT)

// LDS layout (in halves for f16 regions). One WMMA fragment = 32 lanes x 16 halves = 512 halves.
#define ENC_B_H   (48 * 512)   // encoder gate weights: 16 col-tiles x 3 k-tiles (K=96: [Whh|Wih|0])
#define DEC_B_H   (64 * 512)   // decoder gate weights: 16 col-tiles x 4 k-tiles (K=128: [Wih_d|Whh_d])
#define OUT_B_H   (2 * 512)    // W_out: 1 col-tile x 2 k-tiles over h(64)
#define MU_B_H    (8 * 512)    // W_mu / W_lv: {mu,lv} x 4 k-tiles over [h|c](128)
#define H0_B_H    (8 * 512)    // W_l2h / W_l2h2: {h0,c0} x 4 col-tiles, K=16 padded to 32
#define STE_H     (2 * 3 * 512) // encoder A-staging per row-tile: K=96
#define STD_H     (2 * 4 * 512) // decoder/hcat A-staging per row-tile: K=128
#define STM_H     (2 * 512)     // mu A-staging per row-tile: K=16 padded to 32
#define F16_TOTAL (ENC_B_H + DEC_B_H + OUT_B_H + MU_B_H + H0_B_H + STE_H + STD_H + STM_H)
#define MUF32_F   (2 * 16 * 16)
#define HSCR_F    (2 * 16 * 64)
#define SMEM_BYTES (F16_TOTAL * 2 + (MUF32_F + HSCR_F) * 4)  // 159744 B < 320KB WGP LDS

static_assert(SMEM_BYTES <= 160 * 1024, "LDS budget");

// --- CDNA5 WMMA fragment index maps (ISA 7.12.2, wave32) ---
// B (32x16 f16): lane = N + 16*hi ; element e -> k' = (e&7) + ((e&8)<<1) + 8*hi
__device__ __forceinline__ int kmapB(int lane, int e) {
  return (e & 7) + ((e & 8) << 1) + ((lane >> 4) << 3);
}
// staging offset (halves) of A element (row R, column K) within a [kt][32 lanes][16 e] buffer
// A (16x32 f16): lane = M + 16*hi ; element e -> k' = (e&7) + ((e&8)<<1) + 8*hi
__device__ __forceinline__ int stIdx(int K, int R) {
  int kt = K >> 5, kp = K & 31;
  int hi = (kp >> 3) & 1;
  int e  = (kp & 7) | ((kp & 16) >> 1);
  return ((kt << 5) + (hi << 4) + R) * 16 + e;
}

// CDNA5 has a hardware V_TANH_F32 transcendental (co-executes with XDL WMMA).
__device__ __forceinline__ float tanh_fast(float x) {
#if __has_builtin(__builtin_amdgcn_tanhf)
  return __builtin_amdgcn_tanhf(x);
#else
  return tanhf(x);
#endif
}
__device__ __forceinline__ float sigm(float x)  { return 0.5f * tanh_fast(0.5f * x) + 0.5f; }
__device__ __forceinline__ float lrelu(float x) { return x >= 0.f ? x : 0.01f * x; }

__device__ __forceinline__ v8f wmma16(v16h a, v16h b, v8f c) {
  // D = A(16x32,f16) * B(32x16,f16) + C(16x16,f32)
  return __builtin_amdgcn_wmma_f32_16x16x32_f16(false, a, false, b, (short)0, c, false, false);
}

struct Params {
  const float *x, *st, *Wih_e, *Whh_e, *bih_e, *bhh_e, *Wmu, *bmu, *Wlv, *blv,
              *Wl2h, *bl2h, *Wl2h2, *bl2h2, *Wemb, *bemb, *Wih_d, *Whh_d, *bih_d, *bhh_d,
              *Wout, *bout, *Wseq, *bseq, *Wseq2, *bseq2, *Wm1, *bm1, *Wm2, *bm2, *Wm3, *bm3;
  float* out;
};

extern "C" __global__ __launch_bounds__(256)
void vae_lstm_wmma(Params p) {
  extern __shared__ __align__(128) char smem_raw[];
  _Float16* encB = (_Float16*)smem_raw;
  _Float16* decB = encB + ENC_B_H;
  _Float16* outB = decB + DEC_B_H;
  _Float16* muB  = outB + OUT_B_H;
  _Float16* h0B  = muB  + MU_B_H;
  _Float16* stE  = h0B  + H0_B_H;
  _Float16* stD  = stE  + STE_H;
  _Float16* stM  = stD  + STD_H;
  float* muF  = (float*)(encB + F16_TOTAL);   // [2][16][16] f32 mu for heads
  float* hScr = muF + MUF32_F;                // [2][16][64] head scratch

  const int tid  = threadIdx.x;
  const int lane = tid & 31;
  const int warp = tid >> 5;
  const int w    = warp & 3;   // column group (owns h columns 16w..16w+15)
  const int rt   = warp >> 2;  // row-tile (16 batch rows)
  const int n    = lane & 15;
  const int hi   = lane >> 4;
  const int rowbase = blockIdx.x * 32 + rt * 16;

  // ---------------- weight pre-swizzle into WMMA B-fragment layout ----------------
  for (int idx = tid; idx < ENC_B_H; idx += 256) {
    int f = idx >> 9, within = idx & 511, l = within >> 4, e = within & 15;
    int ct = f / 3, kt = f - ct * 3;
    int K = kt * 32 + kmapB(l, e), N = ct * 16 + (l & 15);
    float v = 0.f;
    if (K < 64)      v = p.Whh_e[N * 64 + K];
    else if (K < 72) v = p.Wih_e[N * 8 + (K - 64)];
    encB[idx] = (_Float16)v;
  }
  for (int idx = tid; idx < DEC_B_H; idx += 256) {
    int f = idx >> 9, within = idx & 511, l = within >> 4, e = within & 15;
    int ct = f >> 2, kt = f & 3;
    int K = kt * 32 + kmapB(l, e), N = ct * 16 + (l & 15);
    float v = (K < 64) ? p.Wih_d[N * 64 + K] : p.Whh_d[N * 64 + (K - 64)];
    decB[idx] = (_Float16)v;
  }
  for (int idx = tid; idx < OUT_B_H; idx += 256) {
    int f = idx >> 9, within = idx & 511, l = within >> 4, e = within & 15;
    int K = f * 32 + kmapB(l, e), N = l & 15;
    outB[idx] = (_Float16)((N < 8) ? p.Wout[N * 64 + K] : 0.f);
  }
  for (int idx = tid; idx < MU_B_H; idx += 256) {
    int f = idx >> 9, within = idx & 511, l = within >> 4, e = within & 15;
    int sel = f >> 2, kt = f & 3;
    int K = kt * 32 + kmapB(l, e), N = l & 15;
    muB[idx] = (_Float16)(sel ? p.Wlv[N * 128 + K] : p.Wmu[N * 128 + K]);
  }
  for (int idx = tid; idx < H0_B_H; idx += 256) {
    int f = idx >> 9, within = idx & 511, l = within >> 4, e = within & 15;
    int sel = f >> 2, ct = f & 3;
    int K = kmapB(l, e), N = ct * 16 + (l & 15);
    float v = 0.f;
    if (K < 16) v = sel ? p.Wl2h2[N * 16 + K] : p.Wl2h[N * 16 + K];
    h0B[idx] = (_Float16)v;
  }
  for (int idx = tid; idx < STE_H + STD_H + STM_H; idx += 256) stE[idx] = (_Float16)0.f;

  _Float16* sE = stE + rt * (3 * 512);
  _Float16* sD = stD + rt * (4 * 512);

  // x_0 into encoder A-staging (K=64..71 region)
  if (w == 0) {
    int row = lane >> 1, dp = (lane & 1) * 4;
    const float4 xv = *(const float4*)(p.x + ((size_t)(rowbase + row) * TT) * DD + dp);
    sE[stIdx(64 + dp + 0, row)] = (_Float16)xv.x;
    sE[stIdx(64 + dp + 1, row)] = (_Float16)xv.y;
    sE[stIdx(64 + dp + 2, row)] = (_Float16)xv.z;
    sE[stIdx(64 + dp + 3, row)] = (_Float16)xv.w;
  }
  __syncthreads();

  // per-lane gate biases (torch order i,f,g,o)
  float beg[4], bdg[4];
#pragma unroll
  for (int q = 0; q < 4; ++q) {
    beg[q] = p.bih_e[q * 64 + 16 * w + n] + p.bhh_e[q * 64 + 16 * w + n];
    bdg[q] = p.bih_d[q * 64 + 16 * w + n] + p.bhh_d[q * 64 + 16 * w + n];
  }

  v8f c = {};
  float hvp[8];
#pragma unroll
  for (int j = 0; j < 8; ++j) hvp[j] = 0.f;

  // hoist encoder B fragments into registers (loop-invariant across 512 steps)
  v16h eB[4][3];
#pragma unroll
  for (int q = 0; q < 4; ++q)
#pragma unroll
    for (int kt = 0; kt < 3; ++kt)
      eB[q][kt] = *(const v16h*)(encB + ((q * 4 + w) * 3 + kt) * 512 + lane * 16);

  // ---------------- encoder: 512 recurrent WMMA steps ----------------
  for (int t = 0; t < TT; ++t) {
    v16h a0 = *(const v16h*)(sE + 0 * 512 + lane * 16);
    v16h a1 = *(const v16h*)(sE + 1 * 512 + lane * 16);
    v16h a2 = *(const v16h*)(sE + 2 * 512 + lane * 16);
    v8f acc[4];
#pragma unroll
    for (int q = 0; q < 4; ++q) {
      v8f a = {};
      a = wmma16(a0, eB[q][0], a);
      a = wmma16(a1, eB[q][1], a);
      a = wmma16(a2, eB[q][2], a);
      acc[q] = a;
    }
#pragma unroll
    for (int j = 0; j < 8; ++j) {
      float iv = sigm(acc[0][j] + beg[0]);
      float fv = sigm(acc[1][j] + beg[1]);
      float gv = tanh_fast(acc[2][j] + beg[2]);
      float ov = sigm(acc[3][j] + beg[3]);
      c[j] = fv * c[j] + iv * gv;
      hvp[j] = ov * tanh_fast(c[j]);
    }
    __syncthreads();  // all A-fragment reads complete
#pragma unroll
    for (int j = 0; j < 8; ++j)
      sE[stIdx(16 * w + n, j + 8 * hi)] = (_Float16)hvp[j];
    if (w == 0 && t + 1 < TT) {
      int row = lane >> 1, dp = (lane & 1) * 4;
      const float4 xv = *(const float4*)(p.x + ((size_t)(rowbase + row) * TT + (t + 1)) * DD + dp);
      sE[stIdx(64 + dp + 0, row)] = (_Float16)xv.x;
      sE[stIdx(64 + dp + 1, row)] = (_Float16)xv.y;
      sE[stIdx(64 + dp + 2, row)] = (_Float16)xv.z;
      sE[stIdx(64 + dp + 3, row)] = (_Float16)xv.w;
    }
    __syncthreads();  // writes visible for next step
  }

  // h_cat = [h_T | c_T] into decoder staging (K=128)
#pragma unroll
  for (int j = 0; j < 8; ++j) {
    sD[stIdx(16 * w + n,      j + 8 * hi)] = (_Float16)hvp[j];
    sD[stIdx(64 + 16 * w + n, j + 8 * hi)] = (_Float16)c[j];
  }
  __syncthreads();

  // ---------------- mu / log_var via WMMA ----------------
  if (w < 2) {
    v8f a = {};
#pragma unroll
    for (int kt = 0; kt < 4; ++kt) {
      v16h av = *(const v16h*)(sD + kt * 512 + lane * 16);
      v16h bv = *(const v16h*)(muB + (w * 4 + kt) * 512 + lane * 16);
      a = wmma16(av, bv, a);
    }
    const float bias = (w == 0) ? p.bmu[n] : p.blv[n];
    const size_t off = (w == 0) ? MU_OFF : LV_OFF;
#pragma unroll
    for (int j = 0; j < 8; ++j) {
      int R = j + 8 * hi;
      float v = a[j] + bias;
      p.out[off + (size_t)(rowbase + R) * LL + n] = v;
      if (w == 0) {
        stM[rt * 512 + stIdx(n, R)] = (_Float16)v;
        muF[(rt * 16 + R) * 16 + n] = v;
      }
    }
  }
  __syncthreads();

  // ---------------- decoder init: h0/c0 (WMMA), inp0, heads ----------------
  {
    v16h am = *(const v16h*)(stM + rt * 512 + lane * 16);
    v8f z = {};
    v8f h0a = wmma16(am, *(const v16h*)(h0B + w * 512 + lane * 16), z);
    v8f c0a = wmma16(am, *(const v16h*)(h0B + (4 + w) * 512 + lane * 16), z);
    float bh = p.bl2h[16 * w + n], bc = p.bl2h2[16 * w + n];
#pragma unroll
    for (int j = 0; j < 8; ++j) {
      float hv = lrelu(h0a[j] + bh);
      c[j] = lrelu(c0a[j] + bc);
      hvp[j] = hv;
      sD[stIdx(64 + 16 * w + n, j + 8 * hi)] = (_Float16)hv;
    }
    float v0 = p.bemb[16 * w + n];
#pragma unroll
    for (int d = 0; d < DD; ++d) v0 += p.st[d] * p.Wemb[(16 * w + n) * DD + d];
    _Float16 x0 = (_Float16)fmaxf(v0, 0.f);
#pragma unroll
    for (int j = 0; j < 8; ++j)
      sD[stIdx(16 * w + n, j + 8 * hi)] = x0;
  }
  // num head (scalar, one row per lane)
  if (w == 2 && lane < 16) {
    int R = lane, rowg = rowbase + R;
    const float* muR = &muF[(rt * 16 + R) * 16];
    float acc2 = p.bseq2[0];
    for (int j = 0; j < 64; ++j) {
      float s = p.bseq[j];
      for (int k = 0; k < 16; ++k) s += p.Wseq[j * 16 + k] * muR[k];
      acc2 += p.Wseq2[j] * lrelu(s);
    }
    p.out[NUM_OFF + rowg] = fmaxf(acc2, 0.f);
  }
  // mass head (scalar, one row per lane)
  if (w == 3 && lane < 16) {
    int R = lane, rowg = rowbase + R;
    const float* muR = &muF[(rt * 16 + R) * 16];
    float* a1 = &hScr[(rt * 16 + R) * 64];
    for (int j = 0; j < 64; ++j) {
      float s = p.bm1[j];
      for (int k = 0; k < 16; ++k) s += p.Wm1[j * 16 + k] * muR[k];
      a1[j] = lrelu(s);
    }
    float m0 = p.bm3[0], m1 = p.bm3[1];
    for (int j = 0; j < 64; ++j) {
      float s = p.bm2[j];
      for (int k = 0; k < 64; ++k) s += p.Wm2[j * 64 + k] * a1[k];
      s = lrelu(s);
      m0 += p.Wm3[j] * s;
      m1 += p.Wm3[64 + j] * s;
    }
    float mx = fmaxf(m0, m1);
    float e0 = expf(m0 - mx), e1 = expf(m1 - mx), den = e0 + e1;
    p.out[MASS_OFF + (size_t)rowg * 2 + 0] = e0 / den;
    p.out[MASS_OFF + (size_t)rowg * 2 + 1] = e1 / den;
  }
  __syncthreads();

  // hoist decoder B fragments into registers (loop-invariant across 512 steps)
  v16h dB[4][4];
#pragma unroll
  for (int q = 0; q < 4; ++q)
#pragma unroll
    for (int kt = 0; kt < 4; ++kt)
      dB[q][kt] = *(const v16h*)(decB + ((q * 4 + w) * 4 + kt) * 512 + lane * 16);

  // ---------------- decoder: 512 autoregressive WMMA steps ----------------
  const float bo = (n < 8) ? p.bout[n] : 0.f;
  for (int t = 0; t < TT; ++t) {
    v16h a0 = *(const v16h*)(sD + 0 * 512 + lane * 16);
    v16h a1 = *(const v16h*)(sD + 1 * 512 + lane * 16);
    v16h a2 = *(const v16h*)(sD + 2 * 512 + lane * 16);
    v16h a3 = *(const v16h*)(sD + 3 * 512 + lane * 16);
    v8f acc[4];
#pragma unroll
    for (int q = 0; q < 4; ++q) {
      v8f a = {};
      a = wmma16(a0, dB[q][0], a);
      a = wmma16(a1, dB[q][1], a);
      a = wmma16(a2, dB[q][2], a);
      a = wmma16(a3, dB[q][3], a);
      acc[q] = a;
    }
#pragma unroll
    for (int j = 0; j < 8; ++j) {
      float iv = sigm(acc[0][j] + bdg[0]);
      float fv = sigm(acc[1][j] + bdg[1]);
      float gv = tanh_fast(acc[2][j] + bdg[2]);
      float ov = sigm(acc[3][j] + bdg[3]);
      c[j] = fv * c[j] + iv * gv;
      hvp[j] = ov * tanh_fast(c[j]);
    }
    __syncthreads();  // gate A reads complete
#pragma unroll
    for (int j = 0; j < 8; ++j) {
      sD[stIdx(64 + 16 * w + n, j + 8 * hi)] = (_Float16)hvp[j];          // h_{t+1}
      sD[stIdx(16 * w + n, j + 8 * hi)] = (_Float16)fmaxf(hvp[j], 0.f);   // relu feedback
    }
    __syncthreads();  // h_{t+1} visible
    if (w == 0) {     // y_t = h_{t+1} @ W_out^T + b_out
      v16h h2 = *(const v16h*)(sD + 2 * 512 + lane * 16);
      v16h h3 = *(const v16h*)(sD + 3 * 512 + lane * 16);
      v8f z = {};
      v8f y = wmma16(h2, *(const v16h*)(outB + 0 * 512 + lane * 16), z);
      y = wmma16(h3, *(const v16h*)(outB + 1 * 512 + lane * 16), y);
      if (n < 8) {
#pragma unroll
        for (int j = 0; j < 8; ++j) {
          int R = j + 8 * hi;
          p.out[((size_t)(rowbase + R) * TT + t) * DD + n] = y[j] + bo;
        }
      }
    }
  }
}

extern "C" void kernel_launch(void* const* d_in, const int* in_sizes, int n_in,
                              void* d_out, int out_size, void* d_ws, size_t ws_size,
                              hipStream_t stream) {
  (void)in_sizes; (void)n_in; (void)d_ws; (void)ws_size; (void)out_size;
  Params p;
  p.x      = (const float*)d_in[0];  p.st     = (const float*)d_in[1];
  p.Wih_e  = (const float*)d_in[2];  p.Whh_e  = (const float*)d_in[3];
  p.bih_e  = (const float*)d_in[4];  p.bhh_e  = (const float*)d_in[5];
  p.Wmu    = (const float*)d_in[6];  p.bmu    = (const float*)d_in[7];
  p.Wlv    = (const float*)d_in[8];  p.blv    = (const float*)d_in[9];
  p.Wl2h   = (const float*)d_in[10]; p.bl2h   = (const float*)d_in[11];
  p.Wl2h2  = (const float*)d_in[12]; p.bl2h2  = (const float*)d_in[13];
  p.Wemb   = (const float*)d_in[14]; p.bemb   = (const float*)d_in[15];
  p.Wih_d  = (const float*)d_in[16]; p.Whh_d  = (const float*)d_in[17];
  p.bih_d  = (const float*)d_in[18]; p.bhh_d  = (const float*)d_in[19];
  p.Wout   = (const float*)d_in[20]; p.bout   = (const float*)d_in[21];
  p.Wseq   = (const float*)d_in[22]; p.bseq   = (const float*)d_in[23];
  p.Wseq2  = (const float*)d_in[24]; p.bseq2  = (const float*)d_in[25];
  p.Wm1    = (const float*)d_in[26]; p.bm1    = (const float*)d_in[27];
  p.Wm2    = (const float*)d_in[28]; p.bm2    = (const float*)d_in[29];
  p.Wm3    = (const float*)d_in[30]; p.bm3    = (const float*)d_in[31];
  p.out    = (float*)d_out;

  (void)hipFuncSetAttribute((const void*)vae_lstm_wmma,
                            hipFuncAttributeMaxDynamicSharedMemorySize, SMEM_BYTES);
  vae_lstm_wmma<<<dim3(BT / 32), dim3(256), SMEM_BYTES, stream>>>(p);
}
